// GTOT_23630910062675
// MI455X (gfx1250) — matrix operand order repaired and checked by reading, not compiled, not run on hardware
//
#include <hip/hip_runtime.h>
#include <hip/hip_bf16.h>
#include <math.h>

// Sinkhorn OT for N=4096, EPS=0.1, THRESH=0.1, MAX_ITER=100.
// exp((-C+u_i+v_j)/eps)*mask factorizes as a_i * K_ij * b_j with
// K = exp(-C/eps)*mask precomputed once in bf16 (32MB) + transpose (32MB),
// both L2-resident on MI455X (192MB L2). Each iteration = two matvecs done
// with v_wmma_f32_16x16x32_bf16 using a broadcast vector operand.

#define SK_N 4096
#define SK_EPS 0.1f
#define SK_INV_EPS 10.0f
#define SK_THRESH 0.1f
#define SK_MAX_ITER 100
#define SK_KCHUNK 256   // columns (rows) of K reduced per wave per kernel

typedef __attribute__((ext_vector_type(16))) __bf16 v16bf;
typedef __attribute__((ext_vector_type(8)))  __bf16 v8bf;
typedef __attribute__((ext_vector_type(8)))  float  v8f;

__device__ __forceinline__ v16bf sk_ld16(const __bf16* p0, const __bf16* p1) {
  v8bf lo = *(const v8bf*)p0;
  v8bf hi = *(const v8bf*)p1;
  return __builtin_shufflevector(lo, hi, 0,1,2,3,4,5,6,7,8,9,10,11,12,13,14,15);
}

// ---------------- init ----------------
__global__ __launch_bounds__(256) void sk_init(float* u, float* v, __bf16* a_bf,
                                               __bf16* b_bf, float* s, float* t,
                                               float* err, int* conv, float* cost) {
  int i = blockIdx.x * 256 + threadIdx.x;
  u[i] = 0.0f; v[i] = 0.0f;
  a_bf[i] = (__bf16)1.0f; b_bf[i] = (__bf16)1.0f;
  s[i] = 0.0f; t[i] = 0.0f;
  if (i < 128) err[i] = 0.0f;
  if (i == 0) { *conv = 0; *cost = 0.0f; }
}

// ---------------- build K (bf16), KT (bf16), and copy C to output ----------------
__global__ __launch_bounds__(256) void sk_build_K(const float* __restrict__ C,
                                                  const float* __restrict__ A,
                                                  __bf16* __restrict__ K,
                                                  __bf16* __restrict__ KT,
                                                  float* __restrict__ outC) {
  __shared__ __bf16 tile[32][34];           // +2 pad: conflict-free transpose
  int bi = blockIdx.y, bj = blockIdx.x;
  int tx = threadIdx.x;                     // 0..31
  int ty = threadIdx.y;                     // 0..7
  int r0 = bi * 32, c0 = bj * 32;
  #pragma unroll
  for (int rr = ty; rr < 32; rr += 8) {
    size_t idx = (size_t)(r0 + rr) * SK_N + c0 + tx;
    float c = C[idx];
    float m = A[idx];
    float k = (m > 0.0f) ? __expf(-c * SK_INV_EPS) : 0.0f;
    __bf16 kb = (__bf16)k;
    K[idx] = kb;
    outC[idx] = c;
    tile[rr][tx] = kb;
  }
  __syncthreads();
  #pragma unroll
  for (int rr = ty; rr < 32; rr += 8) {
    KT[(size_t)(c0 + rr) * SK_N + r0 + tx] = tile[tx][rr];
  }
}

// ---------------- rowsum: s_i += sum_j K_ij * b_j  (WMMA bf16) ----------------
// One wave per (16 rows x SK_KCHUNK cols). B operand = b broadcast across 16 cols.
__global__ __launch_bounds__(32) void sk_rowsum(const __bf16* __restrict__ K,
                                                const __bf16* __restrict__ b,
                                                float* __restrict__ s,
                                                const int* __restrict__ conv) {
  if (*conv) return;
  int rt   = blockIdx.y;                    // 0..255 row tile
  int kc   = blockIdx.x;                    // 0..15 k chunk
  int lane = threadIdx.x;
  int M    = lane & 15;
  int half = lane >> 4;
  int i0 = rt * 16;
  int j0 = kc * SK_KCHUNK;
  v8f acc = {};
  const __bf16* rowp = K + (size_t)(i0 + M) * SK_N + j0;
  #pragma unroll 4
  for (int jj = 0; jj < SK_KCHUNK; jj += 32) {
    // A tile (16x32): lane M=row; elems 0..15 -> k=16*half + 0..15
    v16bf av = sk_ld16(rowp + jj + 16 * half, rowp + jj + 16 * half + 8);
    // B tile (32x16) broadcast: lane holds k = 16*half + 0..15, same for all cols
    const __bf16* bp = b + j0 + jj + 16 * half;
    v16bf bv = sk_ld16(bp, bp + 8);
    acc = __builtin_amdgcn_wmma_f32_16x16x32_bf16(false, av, false, bv,
                                                  (short)0, acc, false, false);
  }
  // D layout: lane (N=lane&15, half), VGPR r holds row (r + 8*half). All cols equal.
  if (M == 0) {
    #pragma unroll
    for (int r = 0; r < 8; ++r)
      atomicAdd(&s[i0 + half * 8 + r], acc[r]);
  }
}

// ---------------- colsum: t_j += sum_i a_i * K_ij  (WMMA bf16, via KT) ----------------
// A operand = a broadcast across 16 rows; B tile from KT rows (contiguous loads).
__global__ __launch_bounds__(32) void sk_colsum(const __bf16* __restrict__ KT,
                                                const __bf16* __restrict__ a,
                                                float* __restrict__ t,
                                                const int* __restrict__ conv) {
  if (*conv) return;
  int ct   = blockIdx.y;                    // 0..255 col tile
  int kc   = blockIdx.x;                    // 0..15 k chunk
  int lane = threadIdx.x;
  int nn   = lane & 15;
  int half = lane >> 4;
  int j0 = ct * 16;
  int i0 = kc * SK_KCHUNK;
  v8f acc = {};
  const __bf16* colp = KT + (size_t)(j0 + nn) * SK_N + i0;
  #pragma unroll 4
  for (int ii = 0; ii < SK_KCHUNK; ii += 32) {
    const __bf16* ap = a + i0 + ii;
    v16bf av = sk_ld16(ap + 16 * half, ap + 16 * half + 8);   // broadcast a over rows
    const __bf16* bp = colp + ii + 16 * half;                // B[k][n]=KT[j0+n][i0+ii+k]
    v16bf bv = sk_ld16(bp, bp + 8);
    acc = __builtin_amdgcn_wmma_f32_16x16x32_bf16(false, av, false, bv,
                                                  (short)0, acc, false, false);
  }
  // All rows of D equal; lane's column value is in every acc slot.
  if (half == 0)
    atomicAdd(&t[j0 + nn], acc[0]);
}

// ---------------- u update + err accumulation; zero t for colsum ----------------
__global__ __launch_bounds__(256) void sk_update_u(const float* __restrict__ s,
                                                   float* __restrict__ u,
                                                   __bf16* __restrict__ a_bf,
                                                   float* __restrict__ t,
                                                   float* __restrict__ err_it,
                                                   const int* __restrict__ conv,
                                                   float log_mu) {
  if (*conv) return;
  int i = blockIdx.x * 256 + threadIdx.x;
  float ui  = u[i];
  float row = __expf(ui * SK_INV_EPS) * s[i];
  float un  = SK_EPS * (log_mu - __logf(1e-8f + row)) + ui;
  u[i]    = un;
  a_bf[i] = (__bf16)__expf(un * SK_INV_EPS);
  t[i]    = 0.0f;
  float d = fabsf(un - ui);
  __shared__ float red[256];
  red[threadIdx.x] = d;
  __syncthreads();
  #pragma unroll
  for (int off = 128; off > 0; off >>= 1) {
    if (threadIdx.x < off) red[threadIdx.x] += red[threadIdx.x + off];
    __syncthreads();
  }
  if (threadIdx.x == 0) atomicAdd(err_it, red[0]);
}

// ---------------- v update; zero s for next rowsum ----------------
__global__ __launch_bounds__(256) void sk_update_v(const float* __restrict__ t,
                                                   float* __restrict__ v,
                                                   __bf16* __restrict__ b_bf,
                                                   float* __restrict__ s,
                                                   const int* __restrict__ conv,
                                                   float log_mu) {
  if (*conv) return;
  int j = blockIdx.x * 256 + threadIdx.x;
  float vj  = v[j];
  float col = __expf(vj * SK_INV_EPS) * t[j];
  float vn  = SK_EPS * (log_mu - __logf(1e-8f + col)) + vj;
  v[j]    = vn;
  b_bf[j] = (__bf16)__expf(vn * SK_INV_EPS);
  s[j]    = 0.0f;
}

// ---------------- convergence flag (runs after the full body, like the reference) ----
__global__ void sk_check(const float* __restrict__ err_it, int* __restrict__ conv) {
  if (*err_it < SK_THRESH) *conv = 1;
}

// ---------------- finalize: pi = a_i K b_j recomputed in f32, cost = sum pi*C -------
__global__ __launch_bounds__(256) void sk_finalize(const float* __restrict__ C,
                                                   const float* __restrict__ A,
                                                   const float* __restrict__ u,
                                                   const float* __restrict__ v,
                                                   float* __restrict__ out) {
  size_t idx = (size_t)blockIdx.x * 256 + threadIdx.x;
  int i = (int)(idx >> 12);
  int j = (int)(idx & 4095);
  float c = C[idx];
  float p = (A[idx] > 0.0f) ? __expf((-c + u[i] + v[j]) * SK_INV_EPS) : 0.0f;
  out[1 + idx] = p;
  __shared__ float red[256];
  red[threadIdx.x] = p * c;
  __syncthreads();
  #pragma unroll
  for (int off = 128; off > 0; off >>= 1) {
    if (threadIdx.x < off) red[threadIdx.x] += red[threadIdx.x + off];
    __syncthreads();
  }
  if (threadIdx.x == 0) atomicAdd(&out[0], red[0]);
}

extern "C" void kernel_launch(void* const* d_in, const int* in_sizes, int n_in,
                              void* d_out, int out_size, void* d_ws, size_t ws_size,
                              hipStream_t stream) {
  const int N = SK_N;
  const size_t NN = (size_t)N * N;
  // inputs: x(0), y(1) unused; C(2), A(3)
  const float* C = (const float*)d_in[2];
  const float* A = (const float*)d_in[3];
  float* out = (float*)d_out;            // [0]=cost, [1..NN]=pi, [1+NN..1+2NN]=C

  // workspace layout (needs 64MB + ~100KB)
  char* base = (char*)d_ws;
  __bf16* K  = (__bf16*)base;                       // 32 MB
  __bf16* KT = (__bf16*)(base + NN * 2);            // 32 MB
  float*  u  = (float*)(base + NN * 4);
  float*  v  = u + N;
  float*  s  = v + N;
  float*  t  = s + N;
  float*  err = t + N;                              // 128 slots
  int*    conv = (int*)(err + 128);
  __bf16* a_bf = (__bf16*)(conv + 64);              // 16B aligned
  __bf16* b_bf = a_bf + N;

  const float LOG_MU = logf(1.0f / (float)N + 1e-8f);

  sk_init<<<N / 256, 256, 0, stream>>>(u, v, a_bf, b_bf, s, t, err, conv, out);
  sk_build_K<<<dim3(N / 32, N / 32), dim3(32, 8), 0, stream>>>(C, A, K, KT,
                                                               out + 1 + NN);
  const int kchunks = N / SK_KCHUNK;                // 16
  for (int it = 0; it < SK_MAX_ITER; ++it) {
    sk_rowsum<<<dim3(kchunks, N / 16), 32, 0, stream>>>(K, b_bf, s, conv);
    sk_update_u<<<N / 256, 256, 0, stream>>>(s, u, a_bf, t, err + it, conv, LOG_MU);
    sk_colsum<<<dim3(kchunks, N / 16), 32, 0, stream>>>(KT, a_bf, t, conv);
    sk_update_v<<<N / 256, 256, 0, stream>>>(t, v, b_bf, s, conv, LOG_MU);
    sk_check<<<1, 1, 0, stream>>>(err + it, conv);
  }
  sk_finalize<<<(unsigned)(NN / 256), 256, 0, stream>>>(C, A, u, v, out);
}
